// VQGANImageRepr_85298050499307
// MI455X (gfx1250) — matrix operand order repaired and checked by reading, not compiled, not run on hardware
//
#include <hip/hip_runtime.h>

// ---------------------------------------------------------------------------
// VQGAN nearest-codebook search on gfx1250 (MI455X), wave32 + WMMA bf16.
//   B=256 images, TOK=256 tokens, V=1024 codes, D=256 dims. N = 65536 rows.
//   argmin_v ( ||w_v||^2 - 2 * x . w_v )  with x = (1-a)*W[s1] + a*W[s2]
// Core GEMM X[65536x256] * W^T[256x1024] via v_wmma_f32_16x16x32_bf16.
// Codebook (bf16) double-buffered through LDS with async global->LDS copies
// (ASYNCcnt path), fused running-argmin epilogue, then gather+transpose out.
// ---------------------------------------------------------------------------

typedef __bf16 bf16_t;
typedef __attribute__((ext_vector_type(16))) __bf16 v16bf;
typedef __attribute__((ext_vector_type(8)))  __bf16 v8bf;
typedef __attribute__((ext_vector_type(8)))  float  v8f;
typedef __attribute__((ext_vector_type(4)))  float  v4f;
typedef __attribute__((ext_vector_type(4)))  int    v4i;

#define DIMD 256
#define VCB  1024
#define NTOK 65536

union V16U { v16bf v; v8bf h[2]; };

// Async 16-byte global->LDS copy (gfx1250 GLOBAL_LOAD_ASYNC_TO_LDS_B128).
// Builtin expects v4i pointers: src in addrspace(1), dst in addrspace(3).
__device__ __forceinline__ void async_copy16(const bf16_t* g, bf16_t* l) {
    __builtin_amdgcn_global_load_async_to_lds_b128(
        (__attribute__((address_space(1))) v4i*)g,
        (__attribute__((address_space(3))) v4i*)l,
        0, 0);
}

// ---------------------------------------------------------------------------
// Kernel 1: convert W (f32) -> Wb (bf16, row-major [V,D]) and wn[v] = ||w_v||^2
// One wave per codebook row; 8 rows per 256-thread block -> 128 blocks.
// ---------------------------------------------------------------------------
__global__ void vq_prep_kernel(const float* __restrict__ W,
                               bf16_t* __restrict__ Wb,
                               float* __restrict__ wn) {
    const int lane = threadIdx.x & 31;
    const int wv   = threadIdx.x >> 5;
    const int code = blockIdx.x * 8 + wv;

    const float* src = W  + code * DIMD + lane * 8;
    bf16_t*      dst = Wb + code * DIMD + lane * 8;

    float ss = 0.f;
#pragma unroll
    for (int i = 0; i < 8; ++i) {
        float v = src[i];
        ss += v * v;
        dst[i] = (bf16_t)v;
    }
#pragma unroll
    for (int off = 16; off >= 1; off >>= 1)
        ss += __shfl_xor(ss, off, 32);
    if (lane == 0) wn[code] = ss;
}

// ---------------------------------------------------------------------------
// Kernel 2: fused interpolate + GEMM (WMMA bf16) + running argmin.
//   block = 256 threads = 8 waves; each wave owns 16 tokens (rows).
//   grid.x = 65536 / 128 = 512 blocks.
//   Codebook staged through LDS, 32 codes per chunk, DOUBLE BUFFERED with
//   async global->LDS copies so chunk c+1 fill overlaps chunk c WMMAs.
// ---------------------------------------------------------------------------
#define LDS_STRIDE  264    // 256 bf16 + 8 pad -> 528B rows, bank-spread
#define CHUNK_CODES 32
#define NCHUNKS     (VCB / CHUNK_CODES)

__global__ __launch_bounds__(256, 1)
void vq_search_kernel(const int* __restrict__ seq1,
                      const int* __restrict__ seq2,
                      const float* __restrict__ alpha,
                      const float* __restrict__ W,
                      const bf16_t* __restrict__ Wb,
                      const float* __restrict__ wn,
                      int* __restrict__ idxOut) {
    __shared__ bf16_t ldsB[2 * CHUNK_CODES * LDS_STRIDE];   // 33792 bytes

    const int lane = threadIdx.x & 31;
    const int wave = threadIdx.x >> 5;
    const int h    = lane >> 4;          // lane half (selects K group)
    const int j    = lane & 15;          // row (for A) / col (for B,C)

    const int rowbase = blockIdx.x * 128 + wave * 16;
    const int token   = rowbase + j;     // A-matrix row M == lane%16

    // Async fill of one 32-code chunk: 32 rows x 512B; 8 threads per row,
    // 4 x b128 async copies per thread (16B aligned on both sides).
    auto issue_chunk = [&](int cb, int buf) {
        const int r   = threadIdx.x >> 3;
        const int seg = threadIdx.x & 7;
        const bf16_t* gsrc = Wb + (cb + r) * DIMD + seg * 32;
        bf16_t* ldst = &ldsB[buf * CHUNK_CODES * LDS_STRIDE
                             + r * LDS_STRIDE + seg * 32];
#pragma unroll
        for (int i = 0; i < 4; ++i)
            async_copy16(gsrc + i * 8, ldst + i * 8);
    };

    const float a  = alpha[0];
    const float na = 1.f - a;
    const int s1 = seq1[token];
    const int s2 = seq2[token];

    // Kick off chunk 0 into buffer 0 before building A fragments, so the
    // first fill overlaps the interpolation gathers below.
    issue_chunk(0, 0);

    // --- Build A fragments: 16 tokens x 256 K, bf16, per the ISA 16-bit
    // A layout: lane holds K = k0+8h+e (elems 0..7), k0+16+8h+e (elems 8..15).
    const v4f* W4 = (const v4f*)W;
    v16bf afrag[8];
#pragma unroll
    for (int ks = 0; ks < 8; ++ks) {
        const int k0 = ks * 32;
        const int ka = k0 + 8 * h;        // elems 0..7
        const int kb = k0 + 16 + 8 * h;   // elems 8..15
        v4f p0 = W4[(s1 * DIMD + ka) >> 2];
        v4f p1 = W4[((s1 * DIMD + ka) >> 2) + 1];
        v4f q0 = W4[(s2 * DIMD + ka) >> 2];
        v4f q1 = W4[((s2 * DIMD + ka) >> 2) + 1];
        v4f r0 = W4[(s1 * DIMD + kb) >> 2];
        v4f r1 = W4[((s1 * DIMD + kb) >> 2) + 1];
        v4f t0 = W4[(s2 * DIMD + kb) >> 2];
        v4f t1 = W4[((s2 * DIMD + kb) >> 2) + 1];
#pragma unroll
        for (int e = 0; e < 4; ++e) {
            afrag[ks][e]      = (bf16_t)(na * p0[e] + a * q0[e]);
            afrag[ks][4 + e]  = (bf16_t)(na * p1[e] + a * q1[e]);
            afrag[ks][8 + e]  = (bf16_t)(na * r0[e] + a * t0[e]);
            afrag[ks][12 + e] = (bf16_t)(na * r1[e] + a * t1[e]);
        }
    }

    // Running argmin: lane tracks 8 rows (M = i + 8h), column N = j per tile.
    float minv[8];
    int   mini[8];
#pragma unroll
    for (int i = 0; i < 8; ++i) { minv[i] = 3.4e38f; mini[i] = 0; }

    for (int c = 0; c < NCHUNKS; ++c) {
        const int buf = c & 1;
        const int cb  = c * CHUNK_CODES;

        // My async copies for this chunk must have landed, then barrier so
        // every wave's copies have landed (and everyone is done reading the
        // other buffer, making it safe to refill).
        __builtin_amdgcn_s_wait_asynccnt(0);
        __syncthreads();
        if (c + 1 < NCHUNKS)
            issue_chunk(cb + CHUNK_CODES, buf ^ 1);
        if (c + 2 < NCHUNKS)   // warm GL2 for the chunk after next
            __builtin_prefetch(Wb + (cb + 2 * CHUNK_CODES) * DIMD
                                   + threadIdx.x * 32, 0, 1);

        const bf16_t* lbase = &ldsB[buf * CHUNK_CODES * LDS_STRIDE];
#pragma unroll
        for (int nt = 0; nt < CHUNK_CODES / 16; ++nt) {
            v8f acc = {};
            const int brow = nt * 16 + j;   // B column n within chunk
#pragma unroll
            for (int ks = 0; ks < 8; ++ks) {
                // B fragment: lane holds K = k0+8h+e (elems 0..7), k0+16+8h+e (elems 8..15) of column n.
                const bf16_t* bp = lbase + brow * LDS_STRIDE + ks * 32 + 8 * h;
                V16U bu;
                bu.h[0] = *(const v8bf*)bp;
                bu.h[1] = *(const v8bf*)(bp + 16);
                acc = __builtin_amdgcn_wmma_f32_16x16x32_bf16(
                        false, afrag[ks], false, bu.v,
                        (short)0, acc, false, false);
            }
            const int n  = cb + nt * 16 + j;
            const float w2 = wn[n];
#pragma unroll
            for (int i = 0; i < 8; ++i) {
                float dist = w2 - 2.f * acc[i];
                if (dist < minv[i]) { minv[i] = dist; mini[i] = n; }
            }
        }
    }

    // Cross-lane argmin over the 16 columns in each lane-half (wave32 safe:
    // xor of bits 0..3 never crosses the half boundary). Tie -> smaller idx.
#pragma unroll
    for (int i = 0; i < 8; ++i) {
#pragma unroll
        for (int off = 8; off >= 1; off >>= 1) {
            float ov = __shfl_xor(minv[i], off, 32);
            int   oi = __shfl_xor(mini[i], off, 32);
            if (ov < minv[i] || (ov == minv[i] && oi < mini[i])) {
                minv[i] = ov; mini[i] = oi;
            }
        }
    }
    if (j == 0) {
#pragma unroll
        for (int i = 0; i < 8; ++i)
            idxOut[rowbase + i + 8 * h] = mini[i];
    }
}

// ---------------------------------------------------------------------------
// Kernel 3: z_q[b][d][t] = W[idx[b*256 + t]][d]  (bhwc -> bchw transpose).
//   grid = (256 images, 8 d-chunks of 32), 256 threads = t (coalesced writes;
//   W gathers are L2-resident, W is 1 MB vs 192 MB L2).
// ---------------------------------------------------------------------------
__global__ void vq_gather_kernel(const float* __restrict__ W,
                                 const int* __restrict__ idx,
                                 float* __restrict__ out) {
    __shared__ int s_row[256];
    const int b = blockIdx.x;
    const int dc = blockIdx.y;
    const int t = threadIdx.x;
    s_row[t] = idx[b * 256 + t] * DIMD;
    __syncthreads();
    float* ob = out + (size_t)b * DIMD * 256;
#pragma unroll 4
    for (int d = dc * 32; d < dc * 32 + 32; ++d) {
        ob[(size_t)d * 256 + t] = W[s_row[t] + d];
    }
}

// ---------------------------------------------------------------------------
// Host launcher. Inputs (setup_inputs order): image_seq, image_seq2, alpha, W.
// Workspace: Wb bf16 (512KB) | wn f32 (4KB) | idx i32 (256KB)  = 772KB.
// ---------------------------------------------------------------------------
extern "C" void kernel_launch(void* const* d_in, const int* in_sizes, int n_in,
                              void* d_out, int out_size, void* d_ws, size_t ws_size,
                              hipStream_t stream) {
    const int*   seq1  = (const int*)d_in[0];
    const int*   seq2  = (const int*)d_in[1];
    const float* alpha = (const float*)d_in[2];
    const float* W     = (const float*)d_in[3];
    float* out = (float*)d_out;

    bf16_t* Wb     = (bf16_t*)d_ws;
    float*  wn     = (float*)((char*)d_ws + 524288);
    int*    idxbuf = (int*)((char*)d_ws + 528384);

    vq_prep_kernel<<<VCB / 8, 256, 0, stream>>>(W, Wb, wn);
    vq_search_kernel<<<NTOK / 128, 256, 0, stream>>>(seq1, seq2, alpha, W,
                                                     Wb, wn, idxbuf);
    vq_gather_kernel<<<dim3(256, 8), 256, 0, stream>>>(W, idxbuf, out);
}
